// UResNetSparsePool_63015760166929
// MI455X (gfx1250) — compile-verified, matrix operand order checked by reading, not run on hardware
//
#include <hip/hip_runtime.h>
#include <hip/hip_bf16.h>

typedef __bf16 bf16;
typedef __attribute__((ext_vector_type(8)))  __bf16 v8bf;
typedef __attribute__((ext_vector_type(16))) __bf16 v16bf;
typedef __attribute__((ext_vector_type(8)))  float  v8f;

static __device__ __forceinline__ v8f wmma_bf16(v16bf a, v16bf b, v8f c) {
    // D(16x16,f32) = A(16x32,bf16) x B(32x16,bf16) + C
    return __builtin_amdgcn_wmma_f32_16x16x32_bf16(
        /*neg_a=*/false, a, /*neg_b=*/false, b,
        /*c_mod=*/(short)0, c, /*reuse_a=*/false, /*reuse_b=*/false);
}

// ---------------------------------------------------------------------------
// Generic implicit-GEMM conv (taps = 9 -> 3x3 SAME pad=1; taps = 1 -> 1x1).
// Activations: NHWC bf16 with pixel channel-stride inStride/outStride.
// Weights: wt[n*Ktot + tap*Cin + ci] bf16 (K contiguous per output channel).
// Epilogue: out = relu_opt(acc + bias + residual) * mask_opt  (bf16 store).
// One wave computes a 16(M=pixels) x 16(N=cout) tile.
// ---------------------------------------------------------------------------
__global__ __launch_bounds__(256) void conv_wmma_kernel(
    const bf16* __restrict__ in, int inStride,
    const bf16* __restrict__ wt, const float* __restrict__ bias,
    const bf16* __restrict__ res, int resStride,
    const float* __restrict__ mask,
    bf16* __restrict__ out, int outStride,
    int B, int H, int W, int Cin, int Ktot, int taps, int relu)
{
    int lane = threadIdx.x & 31;
    int wave = threadIdx.x >> 5;
    int M = B * H * W;
    int tile_m = blockIdx.x * 8 + wave;
    if (tile_m * 16 >= M) return;                       // wave-uniform exit
    int n0   = blockIdx.y * 16;
    int row  = lane & 15;
    int half = lane >> 4;
    int klo  = half * 8;

    int m = tile_m * 16 + row;
    bool mval = (m < M);
    int mm = mval ? m : (M - 1);
    int px = mm % W; int t0 = mm / W; int py = t0 % H; int pb = t0 / H;

    const bf16* wrow = wt + (size_t)(n0 + row) * Ktot;
    v8f acc = {};

    for (int t = 0; t < taps; ++t) {
        int dy = (taps == 9) ? (t / 3 - 1) : 0;
        int dx = (taps == 9) ? (t % 3 - 1) : 0;
        int iy = py + dy, ix = px + dx;
        bool v = mval && (iy >= 0) && (iy < H) && (ix >= 0) && (ix < W);
        const bf16* abase = in + ((size_t)(pb * H + iy) * W + ix) * inStride;
        const bf16* bbase = wrow + t * Cin;
        for (int c0 = 0; c0 < Cin; c0 += 32) {
            v8bf z = {};
            v8bf alo = z, ahi = z;
            if (v) {
                alo = *(const v8bf*)(abase + c0 + klo);        // global_load_b128
                ahi = *(const v8bf*)(abase + c0 + klo + 16);
            }
            v8bf blo = *(const v8bf*)(bbase + c0 + klo);
            v8bf bhi = *(const v8bf*)(bbase + c0 + klo + 16);
            v16bf a, b;
            #pragma unroll
            for (int i = 0; i < 8; ++i) {
                a[i] = alo[i]; a[8 + i] = ahi[i];
                b[i] = blo[i]; b[8 + i] = bhi[i];
            }
            acc = wmma_bf16(a, b, acc);
        }
    }

    int n = n0 + (lane & 15);
    float bv = bias ? bias[n] : 0.0f;
    #pragma unroll
    for (int i = 0; i < 8; ++i) {
        int mi = tile_m * 16 + half * 8 + i;
        if (mi >= M) break;
        size_t pix = (size_t)mi;
        float val = acc[i] + bv;
        if (res)  val += (float)res[pix * resStride + n];
        if (relu) val = fmaxf(val, 0.0f);
        if (mask) val *= mask[pix];
        out[pix * outStride + n] = (bf16)val;
    }
}

// ---------------------------------------------------------------------------
// Transposed conv k=2 s=2 as 4 parity-class 1x1 GEMMs over the coarse grid.
// wt4[parity][n][ci] bf16.  out = relu(acc + bias) * mask(fine).
// ---------------------------------------------------------------------------
__global__ __launch_bounds__(256) void upconv_wmma_kernel(
    const bf16* __restrict__ in, int inStride,
    const bf16* __restrict__ wt4, const float* __restrict__ bias,
    const float* __restrict__ mask,
    bf16* __restrict__ out, int outStride,
    int B, int Hc, int Wc, int Cin, int Cout)
{
    int lane = threadIdx.x & 31;
    int wave = threadIdx.x >> 5;
    int M = B * Hc * Wc;
    int tile_m = blockIdx.x * 8 + wave;
    if (tile_m * 16 >= M) return;
    int par = blockIdx.z; int ty = par >> 1, tx = par & 1;
    const bf16* wt = wt4 + (size_t)par * Cout * Cin;

    int n0 = blockIdx.y * 16;
    int row = lane & 15, half = lane >> 4, klo = half * 8;
    int m = tile_m * 16 + row;
    bool mval = (m < M);
    int mm = mval ? m : (M - 1);
    int cx = mm % Wc; int tt = mm / Wc; int cy = tt % Hc; int pb = tt / Hc;

    const bf16* abase = in + ((size_t)(pb * Hc + cy) * Wc + cx) * inStride;
    const bf16* brow  = wt + (size_t)(n0 + row) * Cin;
    v8f acc = {};
    for (int c0 = 0; c0 < Cin; c0 += 32) {
        v8bf z = {};
        v8bf alo = z, ahi = z;
        if (mval) {
            alo = *(const v8bf*)(abase + c0 + klo);
            ahi = *(const v8bf*)(abase + c0 + klo + 16);
        }
        v8bf blo = *(const v8bf*)(brow + c0 + klo);
        v8bf bhi = *(const v8bf*)(brow + c0 + klo + 16);
        v16bf a, b;
        #pragma unroll
        for (int i = 0; i < 8; ++i) {
            a[i] = alo[i]; a[8 + i] = ahi[i];
            b[i] = blo[i]; b[8 + i] = bhi[i];
        }
        acc = wmma_bf16(a, b, acc);
    }

    int n = n0 + (lane & 15);
    float bv = bias[n];
    int Hf = Hc * 2, Wf = Wc * 2;
    #pragma unroll
    for (int i = 0; i < 8; ++i) {
        int mi = tile_m * 16 + half * 8 + i;
        if (mi >= M) break;
        int cxi = mi % Wc; int t2 = mi / Wc; int cyi = t2 % Hc; int bi = t2 / Hc;
        int fy = cyi * 2 + ty, fx = cxi * 2 + tx;
        size_t fpix = (size_t)(bi * Hf + fy) * Wf + fx;
        float val = fmaxf(acc[i] + bv, 0.0f) * mask[fpix];
        out[fpix * outStride + n] = (bf16)val;
    }
}

// ---------------------------------------------------------------------------
// Weight packing: OIHW f32 -> wt[n*K + (ky*KW+kx)*Cin + ci] bf16
// ---------------------------------------------------------------------------
__global__ void pack_conv_w(const float* __restrict__ w, bf16* __restrict__ wt,
                            int Cout, int Cin, int KH, int KW)
{
    int i = blockIdx.x * blockDim.x + threadIdx.x;
    int total = Cout * Cin * KH * KW;
    if (i >= total) return;
    int kx = i % KW; int r = i / KW;
    int ky = r % KH; r /= KH;
    int ci = r % Cin; int n = r / Cin;
    int K = KH * KW * Cin;
    wt[(size_t)n * K + (ky * KW + kx) * Cin + ci] = (bf16)w[i];
}

// IOHW [Cin][Cout][2][2] f32 -> wt4[(ty*2+tx)][n][ci] bf16
__global__ void pack_upconv_w(const float* __restrict__ w, bf16* __restrict__ wt,
                              int Cin, int Cout)
{
    int i = blockIdx.x * blockDim.x + threadIdx.x;
    int total = Cin * Cout * 4;
    if (i >= total) return;
    int tx = i % 2; int r = i / 2;
    int ty = r % 2; r /= 2;
    int n = r % Cout; int ci = r / Cout;
    int p = ty * 2 + tx;
    wt[((size_t)p * Cout + n) * Cin + ci] = (bf16)w[i];
}

// ---------------------------------------------------------------------------
// Masks
// ---------------------------------------------------------------------------
__global__ void mask0_kernel(const float* __restrict__ x, float* __restrict__ m, int N)
{
    int i = blockIdx.x * blockDim.x + threadIdx.x;
    if (i < N) m[i] = (fabsf(x[i]) > 0.0f) ? 1.0f : 0.0f;
}

__global__ void maskpool_kernel(const float* __restrict__ mi, float* __restrict__ mo,
                                int B, int Hc, int Wc)
{
    int idx = blockIdx.x * blockDim.x + threadIdx.x;
    int total = B * Hc * Wc;
    if (idx >= total) return;
    int cx = idx % Wc; int t = idx / Wc; int cy = t % Hc; int b = t / Hc;
    int Wf = Wc * 2;
    size_t base = (size_t)(b * Hc * 2 + cy * 2) * Wf + cx * 2;
    float v = fmaxf(fmaxf(mi[base], mi[base + 1]),
                    fmaxf(mi[base + Wf], mi[base + Wf + 1]));
    mo[idx] = v;
}

// ---------------------------------------------------------------------------
// 2x2/2 max pool on bf16 NHWC (+mask at coarse res). 8 channels per thread.
// ---------------------------------------------------------------------------
__global__ void pool2_kernel(const bf16* __restrict__ in, int inStride,
                             const float* __restrict__ mask,
                             bf16* __restrict__ out, int outStride,
                             int B, int Hc, int Wc, int C)
{
    int idx = blockIdx.x * blockDim.x + threadIdx.x;
    int c8 = C >> 3;
    int total = B * Hc * Wc * c8;
    if (idx >= total) return;
    int c = (idx % c8) * 8; int pix = idx / c8;
    int cx = pix % Wc; int t = pix / Wc; int cy = t % Hc; int b = t / Hc;
    int Hf = Hc * 2, Wf = Wc * 2;
    size_t base = ((size_t)(b * Hf + cy * 2) * Wf + cx * 2) * inStride + c;
    v8bf a = *(const v8bf*)(in + base);
    v8bf b2 = *(const v8bf*)(in + base + inStride);
    v8bf c2 = *(const v8bf*)(in + base + (size_t)Wf * inStride);
    v8bf d2 = *(const v8bf*)(in + base + (size_t)Wf * inStride + inStride);
    float mk = mask[pix];
    v8bf r;
    #pragma unroll
    for (int i = 0; i < 8; ++i) {
        float v = fmaxf(fmaxf((float)a[i], (float)b2[i]),
                        fmaxf((float)c2[i], (float)d2[i]));
        r[i] = (bf16)(v * mk);
    }
    *(v8bf*)(out + (size_t)pix * outStride + c) = r;
}

// ---------------------------------------------------------------------------
// Stem: conv3x3 (1->32) + relu + mask, and 1x1 shortcut (1->32).
// h1 -> NHWC stride 32; sc -> cat0 slice (stride 64).
// ---------------------------------------------------------------------------
__global__ void enc1_front_kernel(const float* __restrict__ x,
    const float* __restrict__ w1, const float* __restrict__ b1,
    const float* __restrict__ ws, const float* __restrict__ bs,
    const float* __restrict__ m0,
    bf16* __restrict__ h1, bf16* __restrict__ sc,
    int B, int H, int W)
{
    int pix = blockIdx.x * blockDim.x + threadIdx.x;
    if (pix >= B * H * W) return;
    int px = pix % W; int t = pix / W; int py = t % H; int b = t / H;
    float xv[9];
    #pragma unroll
    for (int tt = 0; tt < 9; ++tt) {
        int dy = tt / 3 - 1, dx = tt % 3 - 1;
        int iy = py + dy, ix = px + dx;
        xv[tt] = (iy >= 0 && iy < H && ix >= 0 && ix < W)
                 ? x[(size_t)(b * H + iy) * W + ix] : 0.0f;
    }
    float mk = m0[pix];
    float xc = xv[4];
    size_t hb = (size_t)pix * 32, sb = (size_t)pix * 64;
    for (int n = 0; n < 32; ++n) {
        float s = b1[n];
        #pragma unroll
        for (int tt = 0; tt < 9; ++tt) s += xv[tt] * w1[n * 9 + tt];
        h1[hb + n] = (bf16)(fmaxf(s, 0.0f) * mk);
        sc[sb + n] = (bf16)(xc * ws[n] + bs[n]);
    }
}

// ---------------------------------------------------------------------------
// Global average pool over HW for 32 channels (one block per image).
// ---------------------------------------------------------------------------
__global__ __launch_bounds__(256) void avgpool_kernel(const bf16* __restrict__ y,
    int stride, float* __restrict__ pooled, int HW)
{
    __shared__ float sm[256];
    int t = threadIdx.x;
    int c = t & 31;
    int chunk = t >> 5;
    size_t base = (size_t)blockIdx.x * HW;
    float s = 0.0f;
    for (int p = chunk; p < HW; p += 8)
        s += (float)y[(base + p) * stride + c];
    sm[t] = s; __syncthreads();
    if (t < 128) sm[t] += sm[t + 128]; __syncthreads();
    if (t < 64)  sm[t] += sm[t + 64];  __syncthreads();
    if (t < 32) {
        sm[t] += sm[t + 32];
        pooled[blockIdx.x * 32 + t] = sm[t] / (float)HW;
    }
}

__global__ void head_kernel(const float* __restrict__ pooled,
    const float* __restrict__ hw, const float* __restrict__ hb,
    float* __restrict__ outp, int B)
{
    int b = blockIdx.x * blockDim.x + threadIdx.x;
    if (b >= B) return;
    float lg[10];
    float mx = -1e30f;
    for (int j = 0; j < 10; ++j) {
        float s = hb[j];
        for (int c = 0; c < 32; ++c) s += pooled[b * 32 + c] * hw[j * 32 + c];
        lg[j] = s; mx = fmaxf(mx, s);
    }
    float se = 0.0f;
    for (int j = 0; j < 10; ++j) se += expf(lg[j] - mx);
    float lse = mx + logf(se);
    for (int j = 0; j < 10; ++j) outp[b * 10 + j] = lg[j] - lse;
}

// ---------------------------------------------------------------------------
static inline size_t align256(size_t x) { return (x + 255) & ~(size_t)255; }

extern "C" void kernel_launch(void* const* d_in, const int* in_sizes, int n_in,
                              void* d_out, int out_size, void* d_ws, size_t ws_size,
                              hipStream_t stream)
{
    const int B = 2048, H0 = 28, W0 = 28, H1 = 14, W1 = 14, H2 = 7, W2 = 7;
    (void)in_sizes; (void)n_in; (void)out_size;

    // ---- inputs (setup_inputs insertion order, nested dicts depth-first) ----
    int ii = 0;
    const float* x    = (const float*)d_in[ii++];
    const float* e1w1 = (const float*)d_in[ii++]; const float* e1b1 = (const float*)d_in[ii++];
    const float* e1w2 = (const float*)d_in[ii++]; const float* e1b2 = (const float*)d_in[ii++];
    const float* e1ws = (const float*)d_in[ii++]; const float* e1bs = (const float*)d_in[ii++];
    const float* e2w1 = (const float*)d_in[ii++]; const float* e2b1 = (const float*)d_in[ii++];
    const float* e2w2 = (const float*)d_in[ii++]; const float* e2b2 = (const float*)d_in[ii++];
    const float* e2ws = (const float*)d_in[ii++]; const float* e2bs = (const float*)d_in[ii++];
    const float* bw1  = (const float*)d_in[ii++]; const float* bb1  = (const float*)d_in[ii++];
    const float* bw2  = (const float*)d_in[ii++]; const float* bb2  = (const float*)d_in[ii++];
    const float* up1w = (const float*)d_in[ii++]; const float* up1b = (const float*)d_in[ii++];
    const float* d1w1 = (const float*)d_in[ii++]; const float* d1b1 = (const float*)d_in[ii++];
    const float* d1w2 = (const float*)d_in[ii++]; const float* d1b2 = (const float*)d_in[ii++];
    const float* d1ws = (const float*)d_in[ii++]; const float* d1bs = (const float*)d_in[ii++];
    const float* up0w = (const float*)d_in[ii++]; const float* up0b = (const float*)d_in[ii++];
    const float* d0w1 = (const float*)d_in[ii++]; const float* d0b1 = (const float*)d_in[ii++];
    const float* d0w2 = (const float*)d_in[ii++]; const float* d0b2 = (const float*)d_in[ii++];
    const float* d0ws = (const float*)d_in[ii++]; const float* d0bs = (const float*)d_in[ii++];
    const float* hw   = (const float*)d_in[ii++]; const float* hb   = (const float*)d_in[ii++];

    // ---- workspace layout ----
    char* wsb = (char*)d_ws;
    size_t off = 0;
    auto alloc = [&](size_t bytes) -> void* {
        void* p = wsb + off; off = align256(off + bytes); return p;
    };
    float* m0 = (float*)alloc((size_t)B * H0 * W0 * 4);
    float* m1 = (float*)alloc((size_t)B * H1 * W1 * 4);
    float* m2 = (float*)alloc((size_t)B * H2 * W2 * 4);
    bf16* cat0 = (bf16*)alloc((size_t)B * H0 * W0 * 64 * 2);   // [up0 | skip1]
    bf16* cat1 = (bf16*)alloc((size_t)B * H1 * W1 * 128 * 2);  // [up1 | skip2]; reused as y
    bf16* T1 = (bf16*)alloc((size_t)B * H0 * W0 * 32 * 2);
    bf16* T3 = (bf16*)alloc((size_t)B * H1 * W1 * 64 * 2);
    bf16* T4 = (bf16*)alloc((size_t)B * H1 * W1 * 64 * 2);
    bf16* T5 = (bf16*)alloc((size_t)B * H2 * W2 * 64 * 2);
    bf16* T6 = (bf16*)alloc((size_t)B * H2 * W2 * 64 * 2);
    float* pooled = (float*)alloc((size_t)B * 32 * 4);
    bf16* Pe1w2 = (bf16*)alloc(9216 * 2);
    bf16* Pe2w1 = (bf16*)alloc(18432 * 2);
    bf16* Pe2w2 = (bf16*)alloc(36864 * 2);
    bf16* Pe2ws = (bf16*)alloc(2048 * 2);
    bf16* Pbw1  = (bf16*)alloc(36864 * 2);
    bf16* Pbw2  = (bf16*)alloc(36864 * 2);
    bf16* Pup1  = (bf16*)alloc(16384 * 2);
    bf16* Pd1w1 = (bf16*)alloc(73728 * 2);
    bf16* Pd1w2 = (bf16*)alloc(36864 * 2);
    bf16* Pd1ws = (bf16*)alloc(8192 * 2);
    bf16* Pup0  = (bf16*)alloc(8192 * 2);
    bf16* Pd0w1 = (bf16*)alloc(18432 * 2);
    bf16* Pd0w2 = (bf16*)alloc(9216 * 2);
    bf16* Pd0ws = (bf16*)alloc(2048 * 2);
    if (off > ws_size) return;  // insufficient scratch; avoid corruption

    bf16* yfin = cat1;          // cat1 is dead after dec1; reuse as final y [B,28,28,32]

    // ---- weight packing ----
    auto packc = [&](const float* w, bf16* dst, int Cout, int Cin, int KH, int KW) {
        int N = Cout * Cin * KH * KW;
        pack_conv_w<<<(N + 255) / 256, 256, 0, stream>>>(w, dst, Cout, Cin, KH, KW);
    };
    auto packu = [&](const float* w, bf16* dst, int Cin, int Cout) {
        int N = Cin * Cout * 4;
        pack_upconv_w<<<(N + 255) / 256, 256, 0, stream>>>(w, dst, Cin, Cout);
    };
    packc(e1w2, Pe1w2, 32, 32, 3, 3);
    packc(e2w1, Pe2w1, 64, 32, 3, 3);
    packc(e2w2, Pe2w2, 64, 64, 3, 3);
    packc(e2ws, Pe2ws, 64, 32, 1, 1);
    packc(bw1,  Pbw1,  64, 64, 3, 3);
    packc(bw2,  Pbw2,  64, 64, 3, 3);
    packu(up1w, Pup1, 64, 64);
    packc(d1w1, Pd1w1, 64, 128, 3, 3);
    packc(d1w2, Pd1w2, 64, 64, 3, 3);
    packc(d1ws, Pd1ws, 64, 128, 1, 1);
    packu(up0w, Pup0, 64, 32);
    packc(d0w1, Pd0w1, 32, 64, 3, 3);
    packc(d0w2, Pd0w2, 32, 32, 3, 3);
    packc(d0ws, Pd0ws, 32, 64, 1, 1);

    // ---- masks ----
    {
        int N = B * H0 * W0;
        mask0_kernel<<<(N + 255) / 256, 256, 0, stream>>>(x, m0, N);
        int N1 = B * H1 * W1;
        maskpool_kernel<<<(N1 + 255) / 256, 256, 0, stream>>>(m0, m1, B, H1, W1);
        int N2 = B * H2 * W2;
        maskpool_kernel<<<(N2 + 255) / 256, 256, 0, stream>>>(m1, m2, B, H2, W2);
    }

    auto conv = [&](const bf16* in, int inStride, int Cin, int taps,
                    const bf16* wt, const float* bias,
                    const bf16* res, int resStride, const float* mask, int relu,
                    bf16* out, int outStride, int Cout, int H, int W) {
        int M = B * H * W;
        dim3 g((M + 127) / 128, Cout / 16, 1);
        conv_wmma_kernel<<<g, 256, 0, stream>>>(in, inStride, wt, bias, res, resStride,
                                                mask, out, outStride, B, H, W,
                                                Cin, taps * Cin, taps, relu);
    };
    auto upconv = [&](const bf16* in, int inStride, const bf16* wt4, const float* bias,
                      const float* mask, bf16* out, int outStride,
                      int Hc, int Wc, int Cin, int Cout) {
        int M = B * Hc * Wc;
        dim3 g((M + 127) / 128, Cout / 16, 4);
        upconv_wmma_kernel<<<g, 256, 0, stream>>>(in, inStride, wt4, bias, mask,
                                                  out, outStride, B, Hc, Wc, Cin, Cout);
    };
    auto pool = [&](const bf16* in, int inStride, const float* mask,
                    bf16* out, int outStride, int Hc, int Wc, int C) {
        int N = B * Hc * Wc * (C / 8);
        pool2_kernel<<<(N + 255) / 256, 256, 0, stream>>>(in, inStride, mask, out,
                                                          outStride, B, Hc, Wc, C);
    };

    // ---- enc1 : stem + conv2(+sc residual in cat0 slice) -> skip1 ----
    {
        int N = B * H0 * W0;
        enc1_front_kernel<<<(N + 255) / 256, 256, 0, stream>>>(
            x, e1w1, e1b1, e1ws, e1bs, m0, T1, cat0 + 32, B, H0, W0);
    }
    conv(T1, 32, 32, 9, Pe1w2, e1b2, cat0 + 32, 64, m0, 1, cat0 + 32, 64, 32, H0, W0);

    // ---- pool -> enc2 ----
    pool(cat0 + 32, 64, m1, T3, 32, H1, W1, 32);                       // p1
    conv(T3, 32, 32, 9, Pe2w1, e2b1, nullptr, 0, m1, 1, T4, 64, 64, H1, W1);       // h
    conv(T3, 32, 32, 1, Pe2ws, e2bs, nullptr, 0, nullptr, 0, cat1 + 64, 128, 64, H1, W1); // sc
    conv(T4, 64, 64, 9, Pe2w2, e2b2, cat1 + 64, 128, m1, 1, cat1 + 64, 128, 64, H1, W1); // skip2

    // ---- pool -> bottleneck ----
    pool(cat1 + 64, 128, m2, T5, 64, H2, W2, 64);                      // p2
    conv(T5, 64, 64, 9, Pbw1, bb1, nullptr, 0, m2, 1, T6, 64, 64, H2, W2);
    conv(T6, 64, 64, 9, Pbw2, bb2, T5, 64, m2, 1, T3, 64, 64, H2, W2); // bottleneck out

    // ---- up1 -> dec1 ----
    upconv(T3, 64, Pup1, up1b, m1, cat1 + 0, 128, H2, W2, 64, 64);
    conv(cat1, 128, 128, 9, Pd1w1, d1b1, nullptr, 0, m1, 1, T1, 64, 64, H1, W1);
    conv(cat1, 128, 128, 1, Pd1ws, d1bs, nullptr, 0, nullptr, 0, T4, 64, 64, H1, W1);
    conv(T1, 64, 64, 9, Pd1w2, d1b2, T4, 64, m1, 1, T4, 64, 64, H1, W1); // d1 out

    // ---- up0 -> dec0 ----
    upconv(T4, 64, Pup0, up0b, m0, cat0 + 0, 64, H1, W1, 64, 32);
    conv(cat0, 64, 64, 9, Pd0w1, d0b1, nullptr, 0, m0, 1, T1, 32, 32, H0, W0);
    conv(cat0, 64, 64, 1, Pd0ws, d0bs, nullptr, 0, nullptr, 0, yfin, 32, 32, H0, W0);
    conv(T1, 32, 32, 9, Pd0w2, d0b2, yfin, 32, m0, 1, yfin, 32, 32, H0, W0); // y

    // ---- head ----
    avgpool_kernel<<<B, 256, 0, stream>>>(yfin, 32, pooled, H0 * W0);
    head_kernel<<<(B + 255) / 256, 256, 0, stream>>>(pooled, hw, hb, (float*)d_out, B);
}
